// DeformConv2d_28080496181800
// MI455X (gfx1250) — matrix operand order repaired and checked by reading, not compiled, not run on hardware
//
#include <hip/hip_runtime.h>
#include <hip/hip_bf16.h>
#include <stdint.h>

typedef __attribute__((ext_vector_type(16))) __bf16 v16bf;
typedef __attribute__((ext_vector_type(8)))  float  v8f;

#define BATCH   8
#define CIN     256
#define HH      56
#define WWID    56
#define COUT    256
#define KTAPS   9
#define NPIX    (HH*WWID)      // 3136
#define KTOT    (CIN*KTAPS)    // 2304
#define NTILE   64
#define CHUNK_C 32             // channels per K-iteration (one WMMA K=32)
#define KITERS  (KTAPS*(CIN/CHUNK_C))  // 72
#define CSTRIDE 40             // padded LDS row stride (bf16 elems): 80B, 16B-aligned, bank-spread

__device__ __forceinline__ uint32_t f2bf_u(float f) {
    union { float f; uint32_t u; } v; v.f = f;
    return (v.u + 0x7FFFu + ((v.u >> 16) & 1u)) >> 16;   // round-nearest-even
}

// Reorder + convert weight [COUT, CIN, 3, 3] f32  ->  Wb[m, kk*256 + c] bf16
__global__ void dcn_reorder_w(const float* __restrict__ w, uint16_t* __restrict__ wb) {
    int idx = blockIdx.x * blockDim.x + threadIdx.x;
    if (idx >= COUT * KTOT) return;
    int m  = idx / KTOT;
    int k  = idx % KTOT;
    int kk = k / CIN;
    int c  = k % CIN;
    wb[idx] = (uint16_t)f2bf_u(w[(m * CIN + c) * KTAPS + kk]);
}

__global__ __launch_bounds__(256)
void dcn_main(const float* __restrict__ x, const float* __restrict__ offs,
              const float* __restrict__ mask, const uint16_t* __restrict__ wb,
              const float* __restrict__ bias, float* __restrict__ out) {
    // double-buffered column tiles: [buf][n][k] bf16, padded rows
    __shared__ __align__(16) uint16_t colsT[2][NTILE * CSTRIDE];
    __shared__ int   sIdx[KTAPS * NTILE * 4];
    __shared__ float sWgt[KTAPS * NTILE * 4];

    const int tid  = threadIdx.x;
    const int lane = tid & 31;
    const int wv   = tid >> 5;         // wave id 0..7
    const int lhi  = (lane >> 4) & 1;  // lane >= 16
    const int lmod = lane & 15;

    const int b     = blockIdx.x / (NPIX / NTILE);
    const int nbase = (blockIdx.x % (NPIX / NTILE)) * NTILE;

    // ---- Phase 1: per (tap, position) bilinear corners & fused weights ----
    for (int i = tid; i < KTAPS * NTILE; i += 256) {
        int kk = i >> 6;          // 0..8
        int n  = i & 63;
        int ng = nbase + n;
        int ho = ng / WWID, wo = ng % WWID;
        float oy = offs[((b * 2 * KTAPS) + 2 * kk + 0) * NPIX + ng];
        float ox = offs[((b * 2 * KTAPS) + 2 * kk + 1) * NPIX + ng];
        float mv = mask[(b * KTAPS + kk) * NPIX + ng];
        float ys = oy + (float)(ho - 1 + kk / 3);
        float xs = ox + (float)(wo - 1 + kk % 3);
        float y0f = floorf(ys), x0f = floorf(xs);
        int y0 = (int)y0f, x0 = (int)x0f;
        float ly = ys - y0f, lx = xs - x0f;
        float hy = 1.f - ly, hx = 1.f - lx;
        int yc[2] = { y0, y0 + 1 };
        int xc[2] = { x0, x0 + 1 };
        float wy[2] = { hy, ly };
        float wx[2] = { hx, lx };
        #pragma unroll
        for (int j = 0; j < 4; ++j) {
            int yy = yc[j >> 1], xx = xc[j & 1];
            bool valid = (yy >= 0) && (yy < HH) && (xx >= 0) && (xx < WWID);
            int yyc = min(max(yy, 0), HH - 1);
            int xxc = min(max(xx, 0), WWID - 1);
            sIdx[i * 4 + j] = yyc * WWID + xxc;
            sWgt[i * 4 + j] = valid ? (wy[j >> 1] * wx[j & 1] * mv) : 0.f;
        }
    }
    __syncthreads();

    v8f acc[2][4] = {};
    const float* xb0 = x + (size_t)b * CIN * NPIX;

    // gather mapping: thread -> channel pair (c2, c2+1) x 4 positions
    const int c2 = (tid & 15) * 2;   // 0,2,...,30
    const int n0 = tid >> 4;         // 0..15

    // ---- Phase 2: K loop (tap-major, 32 channels/iter), 1 barrier/iter ----
    for (int kit = 0; kit < KITERS; ++kit) {
        const int kk = kit >> 3;   // tap
        const int cg = kit & 7;    // channel group
        const int p  = kit & 1;    // buffer parity

        // gather + pack bf16 pairs into colsT[p] (overlaps WMMAs of iter-1)
        const float* xcA = xb0 + (size_t)(cg * CHUNK_C + c2) * NPIX;
        const float* xcB = xcA + NPIX;
        #pragma unroll
        for (int i = 0; i < 4; ++i) {
            int n = n0 + 16 * i;
            int s = (kk * NTILE + n) * 4;
            int   i0 = sIdx[s + 0], i1 = sIdx[s + 1], i2 = sIdx[s + 2], i3 = sIdx[s + 3];
            float w0 = sWgt[s + 0], w1 = sWgt[s + 1], w2 = sWgt[s + 2], w3 = sWgt[s + 3];
            float vA = w0 * xcA[i0] + w1 * xcA[i1] + w2 * xcA[i2] + w3 * xcA[i3];
            float vB = w0 * xcB[i0] + w1 * xcB[i1] + w2 * xcB[i2] + w3 * xcB[i3];
            uint32_t pk = f2bf_u(vA) | (f2bf_u(vB) << 16);
            *(uint32_t*)&colsT[p][n * CSTRIDE + c2] = pk;
        }

        // A operands (global, LDS-independent): issue before the barrier
        const int kbase = kk * CIN + cg * CHUNK_C;
        v16bf a[2];
        #pragma unroll
        for (int mt = 0; mt < 2; ++mt) {
            int m = wv * 32 + mt * 16 + lmod;
            const uint16_t* ap = wb + (size_t)m * KTOT + kbase + lhi * 8;
            ((uint4*)&a[mt])[0] = *(const uint4*)ap;         // K j..j+7
            ((uint4*)&a[mt])[1] = *(const uint4*)(ap + 16);  // K j+16..j+23
        }

        __syncthreads();   // colsT[p] ready; also retires iter-1's reads of colsT[p^1]

        // B operands + WMMA
        #pragma unroll
        for (int nt = 0; nt < 4; ++nt) {
            const uint16_t* bp = &colsT[p][(nt * 16 + lmod) * CSTRIDE + lhi * 8];
            v16bf bm;
            ((uint4*)&bm)[0] = *(const uint4*)bp;        // K j..j+7
            ((uint4*)&bm)[1] = *(const uint4*)(bp + 16); // K j+16..j+23
            acc[0][nt] = __builtin_amdgcn_wmma_f32_16x16x32_bf16(
                false, a[0], false, bm, (short)0, acc[0][nt], false, false);
            acc[1][nt] = __builtin_amdgcn_wmma_f32_16x16x32_bf16(
                false, a[1], false, bm, (short)0, acc[1][nt], false, false);
        }
    }

    // ---- Epilogue: bias + store (lanes 0..15 coalesced along N) ----
    float* outb = out + (size_t)b * COUT * NPIX;
    #pragma unroll
    for (int mt = 0; mt < 2; ++mt) {
        #pragma unroll
        for (int nt = 0; nt < 4; ++nt) {
            #pragma unroll
            for (int r = 0; r < 8; ++r) {
                int m = wv * 32 + mt * 16 + lhi * 8 + r;
                int n = nbase + nt * 16 + lmod;
                outb[(size_t)m * NPIX + n] = acc[mt][nt][r] + bias[m];
            }
        }
    }
}

extern "C" void kernel_launch(void* const* d_in, const int* in_sizes, int n_in,
                              void* d_out, int out_size, void* d_ws, size_t ws_size,
                              hipStream_t stream) {
    const float* x      = (const float*)d_in[0];
    const float* offset = (const float*)d_in[1];
    const float* mask   = (const float*)d_in[2];
    const float* weight = (const float*)d_in[3];
    const float* bias   = (const float*)d_in[4];
    float* out = (float*)d_out;
    uint16_t* wb = (uint16_t*)d_ws;   // needs COUT*KTOT*2 = 1.18 MB of scratch

    dcn_reorder_w<<<(COUT * KTOT + 255) / 256, 256, 0, stream>>>(weight, wb);
    dcn_main<<<BATCH * (NPIX / NTILE), 256, 0, stream>>>(x, offset, mask, wb, bias, out);
}
